// LM_latent_type_rep_20907900797469
// MI455X (gfx1250) — matrix operand order repaired and checked by reading, not compiled, not run on hardware
//
#include <hip/hip_runtime.h>

// ---- CDNA5 (gfx1250) bf16 WMMA types ----
typedef __attribute__((ext_vector_type(16))) __bf16 v16bf;
typedef __attribute__((ext_vector_type(8)))  __bf16 v8bf;
typedef __attribute__((ext_vector_type(8)))  float  v8f;

union BfFrag { v16bf v; v8bf h8[2]; };

#define B_   16
#define S_   256
#define H_   1024
#define ETOK 512
#define ETAG 128
#define NT_  4
#define TV_  8000
#define KTOT 1664      // ETOK + ETAG + H  (unified A-matrix row)
#define G4H  4096      // 4*H gates
#define OFF_TAG  ETOK          // 512: tag-feedback segment in A row
#define OFF_H    (ETOK + ETAG) // 640: hidden-state segment in A row

__device__ __forceinline__ unsigned short f2bf(float f) {
    unsigned int u = __float_as_uint(f);
    u += 0x7fffu + ((u >> 16) & 1u);          // round-to-nearest-even
    return (unsigned short)(u >> 16);
}
__device__ __forceinline__ float sigmoidf_(float x) { return 1.0f / (1.0f + __expf(-x)); }

#define WMMA_BF16(A, Bm, C) \
    __builtin_amdgcn_wmma_f32_16x16x32_bf16(false, (A), false, (Bm), (short)0, (C), false, false)

// ---------------- prep kernels ----------------

// Abuf[t][b][0..511] = bf16(token_embedding[seq[b][t]][:])
__global__ void k_gather(const int* __restrict__ seq, const float* __restrict__ emb,
                         unsigned short* __restrict__ Abuf) {
    int idx = blockIdx.x * 256 + threadIdx.x;           // < S*B*ETOK
    int e  = idx & (ETOK - 1);
    int tb = idx >> 9;                                  // t*16 + b
    int b  = tb & 15;
    int t  = tb >> 4;
    int tok = seq[b * S_ + t];
    Abuf[(size_t)(t * B_ + b) * KTOT + e] = f2bf(emb[(size_t)tok * ETOK + e]);
}

// Wg[n][k] = bf16 of [W_ih | W_hh] combined, row-major [4096][1664]
__global__ void k_wg(const float* __restrict__ Wih, const float* __restrict__ Whh,
                     unsigned short* __restrict__ Wg) {
    int idx = blockIdx.x * 256 + threadIdx.x;           // < 4096*1664
    int k = idx % KTOT;
    int n = idx / KTOT;
    float v = (k < OFF_H) ? Wih[(size_t)n * OFF_H + k]
                          : Whh[(size_t)n * H_ + (k - OFF_H)];
    Wg[idx] = f2bf(v);
}

// Wp = bf16(W_proj) flat [32000][1024]
__global__ void k_wp(const float* __restrict__ Wproj, unsigned short* __restrict__ Wp) {
    size_t idx = (size_t)blockIdx.x * 256 + threadIdx.x;
    Wp[idx] = f2bf(Wproj[idx]);
}

// Wlhb = bf16(W_lh) [128][1024]
__global__ void k_wlh(const float* __restrict__ Wlh, unsigned short* __restrict__ Wlhb) {
    int idx = blockIdx.x * 256 + threadIdx.x;
    Wlhb[idx] = f2bf(Wlh[idx]);
}

// init: c = 0; slot-0 h segment = 0; slot-0 tagfb = (1/NT) * sum_t W_tag[t]
__global__ void k_init(const float* __restrict__ Wtag, float* __restrict__ cbuf,
                       unsigned short* __restrict__ Abuf) {
    int idx = blockIdx.x * 256 + threadIdx.x;           // 64 blocks -> 16384
    cbuf[idx] = 0.0f;
    {   // h0 = 0 in slot 0
        int b = idx >> 10, j = idx & (H_ - 1);
        Abuf[(size_t)b * KTOT + OFF_H + j] = 0;         // bf16 +0.0
    }
    if (idx < B_ * ETAG) {
        int b = idx >> 7, e = idx & (ETAG - 1);
        float s = 0.0f;
        #pragma unroll
        for (int t = 0; t < NT_; ++t) s += Wtag[t * ETAG + e];
        Abuf[(size_t)b * KTOT + OFF_TAG + e] = f2bf(0.25f * s);
    }
}

// ---------------- per-step kernel 1: gate GEMM via bf16 WMMA ----------------
// gates[16][4096] = Abuf[t] [16][1664] @ Wg^T ; 64 blocks x 4 waves, one 16x16
// N-tile per wave, branch-free k-loop with double-buffered fragments.
__global__ void __launch_bounds__(128) k_gates(const unsigned short* __restrict__ Abase,
                                               const unsigned short* __restrict__ Wg,
                                               float* __restrict__ gates) {
    int lane  = threadIdx.x & 31;
    int wave  = threadIdx.x >> 5;
    int tileN = blockIdx.x * 4 + wave;                  // 0..255
    int m  = lane & 15;
    int hi = lane >> 4;
    v8f acc = {};
    const unsigned short* ar = Abase + (size_t)m * KTOT + hi * 8;
    const unsigned short* br = Wg + (size_t)(tileN * 16 + m) * KTOT + hi * 8;

    BfFrag Ab[2], Bb[2];
    Ab[0].h8[0] = *(const v8bf*)(ar);
    Ab[0].h8[1] = *(const v8bf*)(ar + 16);
    Bb[0].h8[0] = *(const v8bf*)(br);
    Bb[0].h8[1] = *(const v8bf*)(br + 16);
    #pragma unroll 2
    for (int kc = 0; kc < 51; ++kc) {                   // 52 chunks total
        int cur = kc & 1, nxt = cur ^ 1;
        int k1 = (kc + 1) * 32;
        Ab[nxt].h8[0] = *(const v8bf*)(ar + k1);
        Ab[nxt].h8[1] = *(const v8bf*)(ar + k1 + 16);
        Bb[nxt].h8[0] = *(const v8bf*)(br + k1);
        Bb[nxt].h8[1] = *(const v8bf*)(br + k1 + 16);
        acc = WMMA_BF16(Ab[cur].v, Bb[cur].v, acc);
    }
    acc = WMMA_BF16(Ab[1].v, Bb[1].v, acc);             // chunk 51 (51&1 == 1)

    int col = tileN * 16 + m;
    #pragma unroll
    for (int r = 0; r < 8; ++r) {
        int row = r + hi * 8;                            // batch index
        gates[(size_t)row * G4H + col] = acc[r];
    }
}

// ---------------- per-step kernel 2: fused cell + lower proj + tag softmax ----
// 1 block, 1024 threads. Writes h_t and next tag-feedback straight into Abuf
// slot t+1 (the A-matrix of the next step / the word-GEMM input rows).
__global__ void __launch_bounds__(1024) k_celltag(const float* __restrict__ gates,
                                                  float* __restrict__ cbuf,
                                                  unsigned short* __restrict__ Anext,
                                                  const unsigned short* __restrict__ Wlhb,
                                                  const float* __restrict__ blh,
                                                  const float* __restrict__ Wtag,
                                                  const float* __restrict__ btag,
                                                  float* __restrict__ out_tag, int step) {
    __shared__ unsigned short hsh[B_ * H_];    // 32 KB bf16 h
    __shared__ float hl[B_ * ETAG];            // 8 KB h_lower
    __shared__ float tn[B_ * NT_];             // tag logits -> softmax
    int tid = threadIdx.x;

    // Phase A: LSTM cell (thread = hidden unit j, loop over batch)
    #pragma unroll
    for (int b = 0; b < B_; ++b) {
        int idx = b * H_ + tid;
        float gi = gates[(size_t)b * G4H + tid];
        float gf = gates[(size_t)b * G4H + H_ + tid];
        float gg = gates[(size_t)b * G4H + 2 * H_ + tid];
        float go = gates[(size_t)b * G4H + 3 * H_ + tid];
        float c = sigmoidf_(gf) * cbuf[idx] + sigmoidf_(gi) * tanhf(gg);
        float h = sigmoidf_(go) * tanhf(c);
        cbuf[idx] = c;
        unsigned short hb = f2bf(h);
        hsh[idx] = hb;
        Anext[(size_t)b * KTOT + OFF_H + tid] = hb;      // h segment of slot t+1
    }
    __syncthreads();

    // Phase B: h_lower = h @ W_lh^T via WMMA (M=16,N=128,K=1024) on 8 waves
    int lane = tid & 31, wave = tid >> 5;
    if (wave < 8) {
        int m = lane & 15, hi = lane >> 4;
        v8f acc = {};
        const unsigned short* arow = hsh + m * H_ + hi * 8;
        const unsigned short* brow = Wlhb + (size_t)(wave * 16 + m) * H_ + hi * 8;
        BfFrag Ab[2], Bb[2];
        Ab[0].h8[0] = *(const v8bf*)(arow);
        Ab[0].h8[1] = *(const v8bf*)(arow + 16);
        Bb[0].h8[0] = *(const v8bf*)(brow);
        Bb[0].h8[1] = *(const v8bf*)(brow + 16);
        #pragma unroll 2
        for (int kc = 0; kc < 31; ++kc) {               // 32 chunks total
            int cur = kc & 1, nxt = cur ^ 1;
            int k1 = (kc + 1) * 32;
            Ab[nxt].h8[0] = *(const v8bf*)(arow + k1);
            Ab[nxt].h8[1] = *(const v8bf*)(arow + k1 + 16);
            Bb[nxt].h8[0] = *(const v8bf*)(brow + k1);
            Bb[nxt].h8[1] = *(const v8bf*)(brow + k1 + 16);
            acc = WMMA_BF16(Ab[cur].v, Bb[cur].v, acc);
        }
        acc = WMMA_BF16(Ab[1].v, Bb[1].v, acc);          // chunk 31
        int col = wave * 16 + m;
        #pragma unroll
        for (int r = 0; r < 8; ++r) {
            int row = r + hi * 8;
            hl[row * ETAG + col] = acc[r] + blh[col];
        }
    }
    __syncthreads();

    // Phase C1: tag logits [16][4]
    if (tid < B_ * NT_) {
        int b = tid >> 2, t = tid & 3;
        float s = btag[t];
        const float* hv = hl + b * ETAG;
        const float* wv = Wtag + t * ETAG;
        #pragma unroll 8
        for (int e = 0; e < ETAG; ++e) s += hv[e] * wv[e];
        out_tag[((size_t)b * S_ + step) * NT_ + t] = s;
        tn[tid] = s;
    }
    __syncthreads();
    // Phase C2: softmax over 4 tags
    if (tid < B_) {
        float mx = tn[tid * 4];
        for (int t = 1; t < 4; ++t) mx = fmaxf(mx, tn[tid * 4 + t]);
        float ev[4], sum = 0.0f;
        for (int t = 0; t < 4; ++t) { ev[t] = __expf(tn[tid * 4 + t] - mx); sum += ev[t]; }
        for (int t = 0; t < 4; ++t) tn[tid * 4 + t] = ev[t] / sum;
    }
    __syncthreads();
    // Phase C3: next tag-feedback = softmax @ W_tag -> slot t+1 tag segment
    if (tid < B_ * ETAG) {
        int b = tid >> 7, e = tid & (ETAG - 1);
        float s = 0.0f;
        #pragma unroll
        for (int t = 0; t < 4; ++t) s += tn[b * 4 + t] * Wtag[t * ETAG + e];
        Anext[(size_t)b * KTOT + OFF_TAG + e] = f2bf(s);
    }
}

// ---------------- final big GEMM: word logits ----------------
// [4096 x 1024] (all h_t rows inside Abuf, stride KTOT) x [1024 x 32000] + b_proj.
// Each wave: 32x64 macro-tile, fully double-buffered fragments.
__global__ void __launch_bounds__(256) k_word(const unsigned short* __restrict__ HallA,
                                              const unsigned short* __restrict__ Wp,
                                              const float* __restrict__ bproj,
                                              float* __restrict__ out) {
    int lane = threadIdx.x & 31;
    int wave = threadIdx.x >> 5;
    int job  = blockIdx.x * 8 + wave;                   // < 64000
    int mt2 = job / 500;                                // 0..127
    int nt  = job % 500;
    int n0  = nt * 64;
    int m0  = mt2 * 32;
    int m  = lane & 15;
    int hi = lane >> 4;
    v8f acc[8] = {};                                    // [mi*4 + s]
    const unsigned short* ar0 = HallA + (size_t)(m0 + m) * KTOT + hi * 8;
    const unsigned short* ar1 = ar0 + (size_t)16 * KTOT;
    const unsigned short* brb = Wp + (size_t)(n0 + m) * H_ + hi * 8;

    BfFrag A0[2], A1[2], Bv[2][4];
    A0[0].h8[0] = *(const v8bf*)(ar0);
    A0[0].h8[1] = *(const v8bf*)(ar0 + 16);
    A1[0].h8[0] = *(const v8bf*)(ar1);
    A1[0].h8[1] = *(const v8bf*)(ar1 + 16);
    #pragma unroll
    for (int s = 0; s < 4; ++s) {
        const unsigned short* br = brb + (size_t)(s * 16) * H_;
        Bv[0][s].h8[0] = *(const v8bf*)(br);
        Bv[0][s].h8[1] = *(const v8bf*)(br + 16);
    }
    #pragma unroll 2
    for (int kc = 0; kc < 31; ++kc) {                   // 32 chunks total
        int cur = kc & 1, nxt = cur ^ 1;
        int k1 = (kc + 1) * 32;
        A0[nxt].h8[0] = *(const v8bf*)(ar0 + k1);
        A0[nxt].h8[1] = *(const v8bf*)(ar0 + k1 + 16);
        A1[nxt].h8[0] = *(const v8bf*)(ar1 + k1);
        A1[nxt].h8[1] = *(const v8bf*)(ar1 + k1 + 16);
        #pragma unroll
        for (int s = 0; s < 4; ++s) {
            const unsigned short* br = brb + (size_t)(s * 16) * H_ + k1;
            Bv[nxt][s].h8[0] = *(const v8bf*)(br);
            Bv[nxt][s].h8[1] = *(const v8bf*)(br + 16);
        }
        #pragma unroll
        for (int s = 0; s < 4; ++s) {
            acc[s]     = WMMA_BF16(A0[cur].v, Bv[cur][s].v, acc[s]);
            acc[4 + s] = WMMA_BF16(A1[cur].v, Bv[cur][s].v, acc[4 + s]);
        }
    }
    #pragma unroll
    for (int s = 0; s < 4; ++s) {                       // chunk 31 (buffer 1)
        acc[s]     = WMMA_BF16(A0[1].v, Bv[1][s].v, acc[s]);
        acc[4 + s] = WMMA_BF16(A1[1].v, Bv[1][s].v, acc[4 + s]);
    }

    int t0 = m0 >> 4;                                   // timestep of first M-tile
    #pragma unroll
    for (int s = 0; s < 4; ++s) {
        int ncol = n0 + s * 16 + m;
        int tag = ncol / TV_;
        int v   = ncol - tag * TV_;
        float bias = bproj[ncol];
        #pragma unroll
        for (int r = 0; r < 8; ++r) {
            int b = r + hi * 8;                          // batch
            size_t base = ((size_t)tag * B_ + b) * S_;
            out[(base + t0)     * TV_ + v] = acc[s][r]     + bias;
            out[(base + t0 + 1) * TV_ + v] = acc[4 + s][r] + bias;
        }
    }
}

// ---------------- host launch ----------------
extern "C" void kernel_launch(void* const* d_in, const int* in_sizes, int n_in,
                              void* d_out, int out_size, void* d_ws, size_t ws_size,
                              hipStream_t stream) {
    (void)in_sizes; (void)n_in; (void)out_size; (void)ws_size;
    const int*   seq   = (const int*)d_in[0];
    const float* emb   = (const float*)d_in[1];
    const float* Wih   = (const float*)d_in[2];
    const float* Whh   = (const float*)d_in[3];
    const float* Wtag  = (const float*)d_in[4];
    const float* btag  = (const float*)d_in[5];
    const float* Wlh   = (const float*)d_in[6];
    const float* blh   = (const float*)d_in[7];
    const float* Wproj = (const float*)d_in[8];
    const float* bproj = (const float*)d_in[9];

    float* out_tag  = (float*)d_out;                          // [B,S,NT]
    float* out_word = out_tag + (size_t)B_ * S_ * NT_;        // [NT,B,S,TV]

    char* ws = (char*)d_ws;
    size_t off = 0;
    auto alloc = [&](size_t bytes) -> char* {
        char* p = ws + off;
        off += (bytes + 255) & ~(size_t)255;
        return p;
    };
    // Unified A-matrix: (S+1) slots of [B][KTOT] = [x_t | tagfb_t | h_{t-1}]
    unsigned short* Abuf = (unsigned short*)alloc((size_t)(S_ + 1) * B_ * KTOT * 2); // 13.7 MB
    unsigned short* Wg   = (unsigned short*)alloc((size_t)G4H * KTOT * 2);           // 13.6 MB
    unsigned short* Wp   = (unsigned short*)alloc((size_t)NT_ * TV_ * H_ * 2);       // 65.5 MB
    unsigned short* Wlhb = (unsigned short*)alloc((size_t)ETAG * H_ * 2);            // 256 KB
    float*          cbuf  = (float*)alloc((size_t)B_ * H_ * 4);
    float*          gates = (float*)alloc((size_t)B_ * G4H * 4);

    // prep (stream-ordered before the recurrence)
    k_gather<<<(S_ * B_ * ETOK) / 256, 256, 0, stream>>>(seq, emb, Abuf);
    k_wg<<<(G4H * KTOT) / 256, 256, 0, stream>>>(Wih, Whh, Wg);
    k_wp<<<(NT_ * TV_ * H_) / 256, 256, 0, stream>>>(Wproj, Wp);
    k_wlh<<<(ETAG * H_) / 256, 256, 0, stream>>>(Wlh, Wlhb);
    k_init<<<(B_ * H_) / 256, 256, 0, stream>>>(Wtag, cbuf, Abuf);

    // sequential recurrence: 2 kernels per step
    for (int t = 0; t < S_; ++t) {
        k_gates<<<64, 128, 0, stream>>>(Abuf + (size_t)t * B_ * KTOT, Wg, gates);
        k_celltag<<<1, 1024, 0, stream>>>(gates, cbuf,
                                          Abuf + (size_t)(t + 1) * B_ * KTOT,
                                          Wlhb, blh, Wtag, btag, out_tag, t);
    }

    // final batched word-projection GEMM: h_t rows live at Abuf slot t+1, OFF_H
    k_word<<<8000, 256, 0, stream>>>(Abuf + (size_t)B_ * KTOT + OFF_H, Wp, bproj, out_word);
}
